// WordToPhonemeAttention_16398185136642
// MI455X (gfx1250) — compile-verified, matrix operand order checked by reading, not run on hardware
//
#include <hip/hip_runtime.h>
#include <hip/hip_bf16.h>

// ---------------------------------------------------------------------------
// Problem constants (from reference):
//   H=8, DK=DV=64, DM=512, B=8, LQ=512, LK=1024
// Outputs (flat f32, concatenated): out (B*LQ*DM), p, attn_raw, attn_logprob
// ---------------------------------------------------------------------------
#define NEGV   (-1e30f)

typedef __bf16 v16bf __attribute__((ext_vector_type(16)));
typedef float  v8f   __attribute__((ext_vector_type(8)));

// K-index pattern for 16-bit WMMA operand VGPR v (0..7), lane group g (0/1).
// Lane group 0 (lanes 0-15) holds K = {0..7, 16..23}; group 1 holds {8..15, 24..31}.
__device__ __forceinline__ int kpat(int v, int g) {
    return ((v < 4) ? 0 : 16) + ((v & 3) << 1) + (g << 3);
}

// A-operand (16xK tile, M per-lane) from row-major f32, K pairs contiguous.
// Also used for the B-operand when B's K-dim is contiguous per output column
// (e.g. K^T from row-major kh): pass the "column" pointer as the row base.
__device__ __forceinline__ v16bf load_frag_kcontig_f32(const float* __restrict__ base,
                                                       int ld, int lane) {
    v16bf a;
    const int m = lane & 15, g = lane >> 4;
    const float* row = base + (size_t)m * ld;
#pragma unroll
    for (int v = 0; v < 8; ++v) {
        const int k = kpat(v, g);
        const float2 f = *(const float2*)(row + k);
        a[2 * v]     = (__bf16)f.x;
        a[2 * v + 1] = (__bf16)f.y;
    }
    return a;
}

// B-operand (Kx16 tile, N per-lane) from row-major f32 W, K strided by ld.
// base must point at W + k0*ld + n0.
__device__ __forceinline__ v16bf load_frag_kstride_f32(const float* __restrict__ base,
                                                       int ld, int lane) {
    v16bf b;
    const int n = lane & 15, g = lane >> 4;
#pragma unroll
    for (int v = 0; v < 8; ++v) {
        const int k = kpat(v, g);
        b[2 * v]     = (__bf16)base[(size_t)k * ld + n];
        b[2 * v + 1] = (__bf16)base[(size_t)(k + 1) * ld + n];
    }
    return b;
}

// A-operand from LDS bf16 (row-major, K pairs contiguous).
__device__ __forceinline__ v16bf load_frag_lds_bf16(const __bf16* base, int ld, int lane) {
    v16bf a;
    const int m = lane & 15, g = lane >> 4;
    const __bf16* row = base + m * ld;
#pragma unroll
    for (int v = 0; v < 8; ++v) {
        const int k = kpat(v, g);
        a[2 * v]     = row[k];
        a[2 * v + 1] = row[k + 1];
    }
    return a;
}

__device__ __forceinline__ v8f wmma_bf16(v16bf a, v16bf b, v8f c) {
    return __builtin_amdgcn_wmma_f32_16x16x32_bf16(
        /*neg_a=*/false, a, /*neg_b=*/false, b,
        /*c_mod=*/(short)0, c, /*reuse_a=*/false, /*reuse_b=*/false);
}

// ---------------------------------------------------------------------------
// Generic GEMM: Y[M x 512] = X[M x 512] @ W[512 x 512] (+ R), bf16 WMMA, f32 acc.
// grid = (M/16, 4), block = 256 (8 waves); wave w owns 16x16 tile at
// (blockIdx.x*16, blockIdx.y*128 + w*16); K loop of 16 WMMAs.
// Residual add is a compile-time variant to keep the epilogue branch-free.
// ---------------------------------------------------------------------------
template <bool HAS_R>
__global__ __launch_bounds__(256) void gemm512_wmma(const float* __restrict__ X,
                                                    const float* __restrict__ W,
                                                    const float* __restrict__ R,
                                                    float* __restrict__ Y) {
    const int lane = threadIdx.x & 31;
    const int wave = threadIdx.x >> 5;
    const int m0 = blockIdx.x * 16;
    const int n0 = blockIdx.y * 128 + wave * 16;

    const float* xrow = X + (size_t)m0 * 512;
    v8f acc = {};
#pragma unroll 1
    for (int k0 = 0; k0 < 512; k0 += 32) {
        v16bf a = load_frag_kcontig_f32(xrow + k0, 512, lane);
        v16bf b = load_frag_kstride_f32(W + (size_t)k0 * 512 + n0, 512, lane);
        acc = wmma_bf16(a, b, acc);
    }

    const int col = n0 + (lane & 15);
    const int rb  = (lane >> 4) * 8;
#pragma unroll
    for (int r = 0; r < 8; ++r) {
        const size_t idx = (size_t)(m0 + rb + r) * 512 + col;
        float val = acc[r];
        if (HAS_R) val += R[idx];
        Y[idx] = val;
    }
}

// ---------------------------------------------------------------------------
// Prior preprocessing: lprior[b][q][k] = log(prior[b][k][q] + 1e-8).
// LDS 32x32 tile transpose, coalesced on both sides; computed once instead of
// 8x (per-head) scattered gathers in the attention kernel.
// grid = (LK/32, LQ/32, B), block = (32, 8).
// ---------------------------------------------------------------------------
__global__ __launch_bounds__(256) void transpose_log_prior(const float* __restrict__ pr,
                                                           float* __restrict__ lout) {
    __shared__ float tile[32][33];
    const int b  = blockIdx.z;
    const int k0 = blockIdx.x * 32;   // LK tile
    const int q0 = blockIdx.y * 32;   // LQ tile

    const float* src = pr + (size_t)b * 1024 * 512;   // [k][q]
#pragma unroll
    for (int i = threadIdx.y; i < 32; i += 8)
        tile[i][threadIdx.x] = src[(size_t)(k0 + i) * 512 + q0 + threadIdx.x];
    __syncthreads();

    float* dst = lout + (size_t)b * 512 * 1024;       // [q][k]
#pragma unroll
    for (int i = threadIdx.y; i < 32; i += 8)
        dst[(size_t)(q0 + i) * 1024 + k0 + threadIdx.x] =
            __logf(tile[threadIdx.x][i] + 1e-8f);
}

// ---------------------------------------------------------------------------
// Fused attention kernel: one block per (q-tile of 16, b, h).
//   scores = qh . kh^T / 8  -> key mask -> log_softmax + log(prior)
//   -> attn_logprob out; softmax -> *query_mask -> attn_raw out;
//   *mapping_mask -> p out (+ bf16 copy in LDS); O = P @ vh -> oacc ws.
// qh/kh/vh are the projection outputs, row-major (b*L + t)*512 + h*64 + d.
// Large (H,B,LQ,LK) outputs use non-temporal stores (write-once, > L2).
// ---------------------------------------------------------------------------
__global__ __launch_bounds__(256) void attn_fused_wmma(
    const float* __restrict__ qh, const float* __restrict__ kh,
    const float* __restrict__ vh,
    const float* __restrict__ key_mask, const float* __restrict__ query_mask,
    const float* __restrict__ mapping_mask, const float* __restrict__ lprior,
    float* __restrict__ p_out, float* __restrict__ raw_out,
    float* __restrict__ logp_out, float* __restrict__ oacc) {

    __shared__ float  s_scores[16 * 1024];   // 64 KB: scores, then PV partials
    __shared__ __bf16 s_p[16 * 1024];        // 32 KB: p tile in bf16
    __shared__ float  s_red[16][16];
    __shared__ float  s_row[16];

    const int lane = threadIdx.x & 31;
    const int wave = threadIdx.x >> 5;
    const int q0 = blockIdx.x * 16;
    const int b  = blockIdx.y;
    const int h  = blockIdx.z;

    // ---- Phase 1: S = qh_tile (16x64) x kh^T (64x1024), scaled later ----
    const float* qbase = qh + ((size_t)(b * 512 + q0)) * 512 + h * 64;
    const v16bf aq0 = load_frag_kcontig_f32(qbase + 0, 512, lane);
    const v16bf aq1 = load_frag_kcontig_f32(qbase + 32, 512, lane);

#pragma unroll 1
    for (int i = 0; i < 8; ++i) {
        const int n0 = (wave * 8 + i) * 16;
        const float* kb = kh + ((size_t)(b * 1024 + n0)) * 512 + h * 64;
        v16bf b0 = load_frag_kcontig_f32(kb + 0, 512, lane);   // B: K contiguous
        v16bf b1 = load_frag_kcontig_f32(kb + 32, 512, lane);
        v8f acc = {};
        acc = wmma_bf16(aq0, b0, acc);
        acc = wmma_bf16(aq1, b1, acc);
        const int col = n0 + (lane & 15);
        const int rb  = (lane >> 4) * 8;
#pragma unroll
        for (int r = 0; r < 8; ++r) s_scores[(rb + r) * 1024 + col] = acc[r];
    }
    __syncthreads();

    // ---- Phase 2: per-row log-softmax / softmax over LK=1024 ----
    const int row = threadIdx.x >> 4;   // 0..15
    const int sub = threadIdx.x & 15;   // 0..15, column strider
    const int q   = q0 + row;
    float* srow = s_scores + row * 1024;
    const float* km = key_mask + ((size_t)(b * 512 + q)) * 1024;

    // scale + key mask, row max
    float m1 = -3.0e38f;
    for (int j = 0; j < 64; ++j) {
        const int c = sub + j * 16;
        float v = srow[c] * 0.125f;               // 1/sqrt(DK)
        if (km[c] == 0.0f) v = NEGV;
        srow[c] = v;
        m1 = fmaxf(m1, v);
    }
    s_red[row][sub] = m1;
    __syncthreads();
    if (sub == 0) {
        float m = -3.0e38f;
        for (int t = 0; t < 16; ++t) m = fmaxf(m, s_red[row][t]);
        s_row[row] = m;
    }
    __syncthreads();
    m1 = s_row[row];

    float s1 = 0.0f;
    for (int j = 0; j < 64; ++j) s1 += __expf(srow[sub + j * 16] - m1);
    s_red[row][sub] = s1;
    __syncthreads();
    if (sub == 0) {
        float s = 0.0f;
        for (int t = 0; t < 16; ++t) s += s_red[row][t];
        s_row[row] = m1 + __logf(s);              // logZ
    }
    __syncthreads();
    const float logZ = s_row[row];

    // attn = logsoftmax + log(prior + 1e-8); emit attn_logprob; track max
    const float* lpr = lprior + ((size_t)(b * 512 + q)) * 1024;   // contiguous
    float* lp = logp_out + (((size_t)(h * 8 + b)) * 512 + q) * 1024;
    float m2 = -3.0e38f;
    for (int j = 0; j < 64; ++j) {
        const int c = sub + j * 16;
        const float a = (srow[c] - logZ) + lpr[c];
        srow[c] = a;
        __builtin_nontemporal_store(a, &lp[c]);
        m2 = fmaxf(m2, a);
    }
    s_red[row][sub] = m2;
    __syncthreads();
    if (sub == 0) {
        float m = -3.0e38f;
        for (int t = 0; t < 16; ++t) m = fmaxf(m, s_red[row][t]);
        s_row[row] = m;
    }
    __syncthreads();
    m2 = s_row[row];

    float s2 = 0.0f;
    for (int j = 0; j < 64; ++j) s2 += __expf(srow[sub + j * 16] - m2);
    s_red[row][sub] = s2;
    __syncthreads();
    if (sub == 0) {
        float s = 0.0f;
        for (int t = 0; t < 16; ++t) s += s_red[row][t];
        s_row[row] = s;
    }
    __syncthreads();
    const float inv = 1.0f / s_row[row];

    // p_raw / p outputs + bf16 P tile
    const float qm = query_mask[(size_t)b * 512 + q];
    const float* mm = mapping_mask + ((size_t)(b * 512 + q)) * 1024;
    float* raw = raw_out + (((size_t)(h * 8 + b)) * 512 + q) * 1024;
    float* pp  = p_out   + (((size_t)(h * 8 + b)) * 512 + q) * 1024;
    __bf16* prow = s_p + row * 1024;
    for (int j = 0; j < 64; ++j) {
        const int c = sub + j * 16;
        const float praw = __expf(srow[c] - m2) * inv * qm;
        __builtin_nontemporal_store(praw, &raw[c]);
        const float pv = praw * mm[c];
        __builtin_nontemporal_store(pv, &pp[c]);
        prow[c] = (__bf16)pv;
    }
    __syncthreads();

    // ---- Phase 3: O (16x64) = P (16x1024 bf16) @ vh (1024x64), split-K ----
    const int ntile = wave & 3;          // which 16-col slab of DV=64
    const int kbase = (wave >> 2) * 512; // K half per wave pair
    v8f oaccv = {};
#pragma unroll 1
    for (int k0 = 0; k0 < 512; k0 += 32) {
        v16bf a = load_frag_lds_bf16(s_p + kbase + k0, 1024, lane);
        v16bf bf = load_frag_kstride_f32(
            vh + ((size_t)(b * 1024 + kbase + k0)) * 512 + h * 64 + ntile * 16,
            512, lane);
        oaccv = wmma_bf16(a, bf, oaccv);
    }
    const int col = lane & 15;
    const int rb  = (lane >> 4) * 8;
    if (wave >= 4) {
#pragma unroll
        for (int r = 0; r < 8; ++r)
            s_scores[ntile * 256 + (rb + r) * 16 + col] = oaccv[r];
    }
    __syncthreads();
    if (wave < 4) {
        float* ob = oacc + ((size_t)(b * 512 + q0)) * 512 + h * 64 + ntile * 16;
#pragma unroll
        for (int r = 0; r < 8; ++r) {
            const int rr = rb + r;
            ob[(size_t)rr * 512 + col] =
                oaccv[r] + s_scores[ntile * 256 + rr * 16 + col];
        }
    }
}

// ---------------------------------------------------------------------------
// Launch: prior transpose -> 3 projection GEMMs -> fused attention -> FC GEMM.
// Workspace (f32): qh 2M | kh 4M | vh 4M | oacc 2M | lprior 4M = 64 MB.
// d_out (f32): out 2,097,152 | p 33,554,432 | raw 33,554,432 | logp 33,554,432.
// ---------------------------------------------------------------------------
extern "C" void kernel_launch(void* const* d_in, const int* in_sizes, int n_in,
                              void* d_out, int out_size, void* d_ws, size_t ws_size,
                              hipStream_t stream) {
    const float* q            = (const float*)d_in[0];
    const float* k            = (const float*)d_in[1];
    const float* v            = (const float*)d_in[2];
    const float* w_qs         = (const float*)d_in[3];
    const float* w_ks         = (const float*)d_in[4];
    const float* w_vs         = (const float*)d_in[5];
    const float* w_fc         = (const float*)d_in[6];
    const float* key_mask     = (const float*)d_in[7];
    const float* query_mask   = (const float*)d_in[8];
    const float* mapping_mask = (const float*)d_in[9];
    const float* attn_prior   = (const float*)d_in[10];

    float* ws     = (float*)d_ws;
    float* qh     = ws;                 // 8*512*512   = 2,097,152
    float* kh     = ws + 2097152;       // 8*1024*512  = 4,194,304
    float* vh     = ws + 6291456;       // 8*1024*512  = 4,194,304
    float* oacc   = ws + 10485760;      // 8*512*512   = 2,097,152
    float* lprior = ws + 12582912;      // 8*512*1024  = 4,194,304

    float* out      = (float*)d_out;
    float* p_out    = out + 2097152;
    float* raw_out  = out + 35651584;
    float* logp_out = out + 69206016;

    // log(prior + 1e-8), transposed to (B, LQ, LK)
    transpose_log_prior<<<dim3(32, 16, 8), dim3(32, 8), 0, stream>>>(attn_prior, lprior);

    // Projections: qh = q @ w_qs (M=4096), kh = k @ w_ks, vh = v @ w_vs (M=8192)
    gemm512_wmma<false><<<dim3(256, 4), 256, 0, stream>>>(q, w_qs, nullptr, qh);
    gemm512_wmma<false><<<dim3(512, 4), 256, 0, stream>>>(k, w_ks, nullptr, kh);
    gemm512_wmma<false><<<dim3(512, 4), 256, 0, stream>>>(v, w_vs, nullptr, vh);

    // Fused attention: grid = (LQ/16, B, H)
    attn_fused_wmma<<<dim3(32, 8, 8), 256, 0, stream>>>(
        qh, kh, vh, key_mask, query_mask, mapping_mask, lprior,
        p_out, raw_out, logp_out, oacc);

    // Final: out = oacc @ w_fc + q (residual)
    gemm512_wmma<true><<<dim3(256, 4), 256, 0, stream>>>(oacc, w_fc, q, out);
}
